// FGKAN_48584670052950
// MI455X (gfx1250) — matrix operand order repaired and checked by reading, not compiled, not run on hardware
//
#include <hip/hip_runtime.h>

#define B_SZ 4096
#define T_SZ 32
#define EDIM 64
#define EDIM2 128

typedef __attribute__((ext_vector_type(16))) _Float16 v16h;
typedef __attribute__((ext_vector_type(8)))  _Float16 v8h;
typedef __attribute__((ext_vector_type(8)))  float    v8f;

// Fast sigmoid: v_exp_f32 + v_rcp_f32 (no IEEE div expansion).
__device__ __forceinline__ float sigmoidf(float x) {
    return __builtin_amdgcn_rcpf(1.0f + __expf(-x));
}

// ---------------------------------------------------------------------------
// Pre-swizzle W1 (f32 [k=128][n=64]) into per-lane WMMA B-fragment order:
// W1s[(((s*4 + nt)*32 + lane)*16 + idx)] holds the half that lane `lane`
// needs at slot `idx` for K-step s, N-tile nt:
//   hgrp = lane>>4, n = nt*16 + (lane&15), k = s*32 + hgrp*16 + idx
// so each lane reads one contiguous 32-byte run per fragment.
// ---------------------------------------------------------------------------
__global__ void w1_swizzle_kernel(const float* __restrict__ W1,
                                  _Float16* __restrict__ W1s) {
    int o = blockIdx.x * blockDim.x + threadIdx.x;
    if (o >= EDIM2 * EDIM) return;
    int idx  = o & 15;
    int lane = (o >> 4) & 31;
    int nt   = (o >> 9) & 3;
    int s    = o >> 11;
    int n = nt * 16 + (lane & 15);
    int k = s * 32 + (lane >> 4) * 16 + idx;
    W1s[o] = (_Float16)W1[k * EDIM + n];
}

// ---------------------------------------------------------------------------
// Fused per-(branch,layer) attention kernel.
// Block = 1 batch element, 256 threads = 8 waves (wave32).
// gather -> LDS(f16) -> WMMA GEMM [32x128]x[128x64] -> sigmoid,*W2 -> row sum
// -> sigmoid -> softmax(T) -> weighted sum of t_emb -> accumulate out[b][64]
// ---------------------------------------------------------------------------
__global__ __launch_bounds__(256)
void attn_kernel(const float* __restrict__ ent, const float* __restrict__ rel,
                 const int* __restrict__ hidx, const int* __restrict__ ridx,
                 const int* __restrict__ tidx,
                 const _Float16* __restrict__ W1s, const float* __restrict__ W2,
                 float* __restrict__ out, int layer)
{
    __shared__ _Float16 sW1[EDIM2 * EDIM];   // swizzled fragments, 16 KB
    __shared__ _Float16 sX[T_SZ * EDIM2];    // [t][k]  8 KB
    __shared__ float    sT[T_SZ * EDIM];     // t_emb   8 KB
    __shared__ float    sProd[T_SZ * EDIM];  // sig(y)*W2[n] 8 KB
    __shared__ float    sW2[EDIM];
    __shared__ float    sA[T_SZ];

    const int b   = blockIdx.x;
    const int tid = threadIdx.x;

    // ---- stage swizzled W1 (f16) into LDS as dword copies ----------------
    {
        const unsigned* src = (const unsigned*)W1s;
        unsigned*       dst = (unsigned*)sW1;
        for (int i = tid; i < EDIM2 * EDIM / 2; i += 256) dst[i] = src[i];
    }
    if (tid < EDIM) sW2[tid] = W2[tid];

    // ---- gather x = [h_set | path] (f16) and t_emb (f32) -----------------
    {
        const int  r    = tid >> 3;            // triple 0..31
        const int  c0   = (tid & 7) * 8;       // dim chunk
        const long off0 = (long)b * T_SZ + r;                  // layer-0 slice
        const long offL = (long)layer * B_SZ * T_SZ + off0;    // layer-i slice
        const float* eh0 = ent + (long)hidx[off0] * EDIM;
        const float* rr0 = rel + (long)ridx[off0] * EDIM;
        const float* te  = ent + (long)tidx[offL] * EDIM;
        if (layer == 0) {
            for (int d = 0; d < 8; ++d) {
                sX[r * EDIM2 + c0 + d]        = (_Float16)eh0[c0 + d];
                sX[r * EDIM2 + EDIM + c0 + d] = (_Float16)rr0[c0 + d];
                sT[r * EDIM + c0 + d]         = te[c0 + d];
            }
        } else {
            const float* eh1 = ent + (long)hidx[offL] * EDIM;
            const float* rr1 = rel + (long)ridx[offL] * EDIM;
            for (int d = 0; d < 8; ++d) {
                sX[r * EDIM2 + c0 + d]        = (_Float16)(eh0[c0 + d] + eh1[c0 + d]);
                sX[r * EDIM2 + EDIM + c0 + d] = (_Float16)(rr0[c0 + d] * rr1[c0 + d]);
                sT[r * EDIM + c0 + d]         = te[c0 + d];
            }
        }
    }
    __syncthreads();

    // ---- WMMA GEMM: Y[32x64] = X[32x128] * W1[128x64] --------------------
    {
        const int wave = tid >> 5;
        const int lane = tid & 31;
        const int m0   = (wave >> 2) * 16;     // M tile (0|16)
        const int nt   = wave & 3;             // N tile index
        const int n0   = nt * 16;
        const int hgrp = lane >> 4;            // half-wave 0|1
        const int mrow = m0 + (lane & 15);     // A: lane = M row
        const int ncol = n0 + (lane & 15);     // B/C: lane = N col
        v8f c = {};
        #pragma unroll
        for (int s = 0; s < 4; ++s) {          // K = 4 x 32
            const int kb = s * 32;
            // A fragment: two contiguous 8-half runs per lane (16B aligned)
            const v8h a_lo = *(const v8h*)&sX[mrow * EDIM2 + kb + hgrp * 16];
            const v8h a_hi = *(const v8h*)&sX[mrow * EDIM2 + kb + hgrp * 16 + 8];
            const v16h a = __builtin_shufflevector(a_lo, a_hi,
                    0, 1, 2, 3, 4, 5, 6, 7, 8, 9, 10, 11, 12, 13, 14, 15);
            // B fragment: one contiguous 32-byte run per lane (pre-swizzled)
            const v16h w = *(const v16h*)&sW1[((s * 4 + nt) * 32 + lane) * 16];
            c = __builtin_amdgcn_wmma_f32_16x16x32_f16(
                    false, a, false, w, (short)0, c, false, false);
        }
        // C layout: lane = N, VGPR v -> M = v + 8*hgrp. Fuse sigmoid * W2[n].
        const float w2n = sW2[ncol];
        #pragma unroll
        for (int v = 0; v < 8; ++v) {
            int M = m0 + v + 8 * hgrp;
            sProd[M * EDIM + ncol] = sigmoidf(c[v]) * w2n;
        }
    }
    __syncthreads();

    // ---- logits, second sigmoid, softmax over T (wave 0, 1 lane per t) ---
    if (tid < T_SZ) {
        float s = 0.f;
        const float4* row = (const float4*)&sProd[tid * EDIM];
        #pragma unroll
        for (int n = 0; n < EDIM / 4; ++n) {
            float4 p = row[n];
            s += p.x + p.y + p.z + p.w;
        }
        float aval = sigmoidf(s);
        float m = aval;
        #pragma unroll
        for (int off = 16; off > 0; off >>= 1)
            m = fmaxf(m, __shfl_xor(m, off, 32));
        float e = __expf(aval - m);
        float se = e;
        #pragma unroll
        for (int off = 16; off > 0; off >>= 1)
            se += __shfl_xor(se, off, 32);
        sA[tid] = e * __builtin_amdgcn_rcpf(se);
    }
    __syncthreads();

    // ---- weighted sum of t_emb; accumulate across layers -----------------
    if (tid < EDIM) {
        float acc = 0.f;
        #pragma unroll
        for (int t = 0; t < T_SZ; ++t) acc += sA[t] * sT[t * EDIM + tid];
        float* o = out + (long)b * EDIM + tid;
        if (layer == 0) *o = acc;              // init (ws is poisoned)
        else            *o += acc;             // stream-ordered accumulate
    }
}

// ---------------------------------------------------------------------------
// Final combine: means + sums + concat-dot + sigmoid -> scores[B], loss=0
// ---------------------------------------------------------------------------
__device__ __forceinline__ float mean_gather(const float* __restrict__ ent,
                                             const int* __restrict__ row, int d) {
    float s = 0.f;
    for (int t = 0; t < T_SZ; ++t) s += ent[(long)row[t] * EDIM + d];
    return s * (1.0f / (float)T_SZ);
}

__global__ __launch_bounds__(64)
void final_kernel(const int* __restrict__ items,
                  const int* __restrict__ uis_h, const int* __restrict__ ipt_h,
                  const int* __restrict__ upt_h, const int* __restrict__ iot_h,
                  const float* __restrict__ ent,
                  const float* __restrict__ uis_o, const float* __restrict__ ipt_o,
                  const float* __restrict__ upt_o, const float* __restrict__ iot_o,
                  float* __restrict__ d_out)
{
    __shared__ float red[EDIM];
    const int  b  = blockIdx.x;
    const int  d  = threadIdx.x;
    const long bo = (long)b * EDIM + d;
    const long ro = (long)b * T_SZ;   // layer-0 slice of [L,B,T]

    float e_u  = mean_gather(ent, uis_h + ro, d) + uis_o[bo];
    float e_i  = ent[(long)items[b] * EDIM + d] + ipt_o[bo]
               + mean_gather(ent, ipt_h + ro, d);           // extra 'sum' term
    float e_pu = mean_gather(ent, upt_h + ro, d) + upt_o[bo];
    float e_pi = mean_gather(ent, iot_h + ro, d) + iot_o[bo];

    // cat([e_u,e_pu]) . cat([e_pi,e_i]) == e_pi.e_u + e_i.e_pu
    red[d] = e_pi * e_u + e_i * e_pu;
    __syncthreads();
    for (int off = 32; off > 0; off >>= 1) {
        if (d < off) red[d] += red[d + off];
        __syncthreads();
    }
    if (d == 0)           d_out[b]    = sigmoidf(red[0]);
    if (b == 0 && d == 0) d_out[B_SZ] = 0.0f;   // kge_loss
}

// ---------------------------------------------------------------------------
extern "C" void kernel_launch(void* const* d_in, const int* in_sizes, int n_in,
                              void* d_out, int out_size, void* d_ws, size_t ws_size,
                              hipStream_t stream)
{
    (void)in_sizes; (void)n_in; (void)out_size; (void)ws_size;

    const int*   items = (const int*)d_in[0];
    const int*   hs[4] = {(const int*)d_in[1], (const int*)d_in[4],
                          (const int*)d_in[7], (const int*)d_in[10]};
    const int*   rs[4] = {(const int*)d_in[2], (const int*)d_in[5],
                          (const int*)d_in[8], (const int*)d_in[11]};
    const int*   ts[4] = {(const int*)d_in[3], (const int*)d_in[6],
                          (const int*)d_in[9], (const int*)d_in[12]};
    const float* ent = (const float*)d_in[13];
    const float* rel = (const float*)d_in[14];
    const float* W1  = (const float*)d_in[15];
    const float* W2  = (const float*)d_in[16];

    _Float16* W1s   = (_Float16*)d_ws;                       // 16 KB swizzled
    float*    slab0 = (float*)((char*)d_ws + 16384);         // 4 x 1 MB
    float*    outs[4];
    for (int i = 0; i < 4; ++i) outs[i] = slab0 + (size_t)i * B_SZ * EDIM;

    w1_swizzle_kernel<<<(EDIM2 * EDIM + 255) / 256, 256, 0, stream>>>(W1, W1s);

    for (int br = 0; br < 4; ++br)
        for (int l = 0; l < 2; ++l)
            attn_kernel<<<B_SZ, 256, 0, stream>>>(ent, rel, hs[br], rs[br],
                                                  ts[br], W1s, W2, outs[br], l);

    final_kernel<<<B_SZ, 64, 0, stream>>>(items, hs[0], hs[1], hs[2], hs[3], ent,
                                          outs[0], outs[1], outs[2], outs[3],
                                          (float*)d_out);
}
